// CoAttention_75874892251202
// MI455X (gfx1250) — matrix operand order repaired and checked by reading, not compiled, not run on hardware
//
#include <hip/hip_runtime.h>
#include <hip/hip_bf16.h>
#include <math.h>

#define N_NODES 8191
#define NPAD    8192
#define H       256
#define MDIM    256
#define GDIM    1280   /* 5*M  */
#define XDIM    768    /* 3*H  */
#define KCAT    1280   /* 768 + 256 + 256 */

typedef __attribute__((ext_vector_type(16))) __bf16 v16bf;
typedef __attribute__((ext_vector_type(8)))  __bf16 v8bf;
typedef __attribute__((ext_vector_type(8)))  float  v8f;

static __device__ __forceinline__ v8f vzero8() {
    v8f v;
#pragma unroll
    for (int i = 0; i < 8; ++i) v[i] = 0.0f;
    return v;
}

static __device__ __forceinline__ v16bf joinbf(v8bf a, v8bf b) {
    v16bf r;
#pragma unroll
    for (int i = 0; i < 8; ++i) { r[i] = a[i]; r[8 + i] = b[i]; }
    return r;
}

// A-matrix fragment (16x32 bf16): lane row = lane&15, hi = lane>>4.
// v0..3 = K[k0 + hi*8 .. +7], v4..7 = K[k0 + 16 + hi*8 .. +7]  (per ISA 7.12.2)
static __device__ __forceinline__ v16bf load_frag_a(const __bf16* base, int row,
                                                    int stride, int k0, int hi) {
    const __bf16* p = base + (size_t)row * stride + k0 + hi * 8;
    v8bf lo = *(const v8bf*)(p);
    v8bf hv = *(const v8bf*)(p + 16);
    return joinbf(lo, hv);
}

// B-matrix fragment (32x16 bf16): lane col = lane&15, hi = lane>>4.
// v0..3 = K[k0 + hi*8 .. +7], v4..7 = K[k0 + 16 + hi*8 .. +7]  (same k order as A)
static __device__ __forceinline__ v16bf load_frag_b(const __bf16* base, int idx,
                                                    int stride, int k0, int hi) {
    const __bf16* p = base + (size_t)idx * stride + k0 + hi * 8;
    v8bf lo = *(const v8bf*)(p);
    v8bf hv = *(const v8bf*)(p + 16);
    return joinbf(lo, hv);
}

static __device__ __forceinline__ v8f wmma_bf16(v16bf a, v16bf b, v8f c) {
    return __builtin_amdgcn_wmma_f32_16x16x32_bf16(false, a, false, b,
                                                   (short)0, c, false, false);
}

// ---------------------------------------------------------------------------
// K1a: cast hidden (8191x256 f32) -> Hb (8192x256 bf16, zero pad) and HbT
// ---------------------------------------------------------------------------
__global__ __launch_bounds__(256) void cast_hidden_kernel(
    const float* __restrict__ hidden, __bf16* __restrict__ Hb,
    __bf16* __restrict__ HbT) {
    int idx = blockIdx.x * blockDim.x + threadIdx.x;
    if (idx >= NPAD * H) return;
    int r = idx / H, c = idx % H;
    float v = (r < N_NODES) ? hidden[(size_t)r * H + c] : 0.0f;
    __bf16 bv = (__bf16)v;
    Hb[idx] = bv;
    HbT[(size_t)c * NPAD + r] = bv;
}

// ---------------------------------------------------------------------------
// K1b: WcatT[n][k] = Wcat[k][n], Wcat = [W ; U_l ; U_r]  (1280 x 1280), bf16
// ---------------------------------------------------------------------------
__global__ __launch_bounds__(256) void build_wcatT_kernel(
    const float* __restrict__ W, const float* __restrict__ Ul,
    const float* __restrict__ Ur, __bf16* __restrict__ WcatT) {
    int idx = blockIdx.x * blockDim.x + threadIdx.x;
    if (idx >= GDIM * KCAT) return;
    int col = idx / KCAT;
    int k   = idx % KCAT;
    float v;
    if (k < XDIM)             v = W[(size_t)k * GDIM + col];
    else if (k < XDIM + MDIM) v = Ul[(size_t)(k - XDIM) * GDIM + col];
    else                      v = Ur[(size_t)(k - XDIM - MDIM) * GDIM + col];
    WcatT[(size_t)col * KCAT + k] = (__bf16)v;
}

// ---------------------------------------------------------------------------
// K2: fused flash attention + x = [h, h-enc, h*enc] epilogue (writes Xb bf16)
// One wave = 16 query rows x 128 output cols (half). 4 waves / block.
// ---------------------------------------------------------------------------
__global__ __launch_bounds__(128) void attn_kernel(
    const float* __restrict__ hidden, const __bf16* __restrict__ Hb,
    const __bf16* __restrict__ HbT, __bf16* __restrict__ Xb) {
    __shared__ __bf16 Plds[4][16 * 32];

    const int lane = threadIdx.x & 31;
    const int wave = threadIdx.x >> 5;
    const int wid  = blockIdx.x * 4 + wave;   // 0..1023
    const int qtile = wid >> 1;               // 0..511
    const int half  = wid & 1;                // output column half
    const int nn = lane & 15;
    const int hi = lane >> 4;
    const int qrow = qtile * 16;
    __bf16* P = Plds[wave];

    // resident Q fragments (A) for the 8 K-chunks of H=256
    v16bf qf[8];
#pragma unroll
    for (int kc = 0; kc < 8; ++kc)
        qf[kc] = load_frag_a(Hb, qrow + nn, H, kc * 32, hi);

    v8f O[8];
#pragma unroll
    for (int i = 0; i < 8; ++i) O[i] = vzero8();
    float m[8], l[8];
#pragma unroll
    for (int j = 0; j < 8; ++j) { m[j] = -__builtin_inff(); l[j] = 0.0f; }

    for (int kb = 0; kb < NPAD; kb += 32) {
        // S = Q @ K^T for 32 keys (two 16x16 tiles)
        v8f s0 = vzero8(), s1 = vzero8();
#pragma unroll
        for (int kc = 0; kc < 8; ++kc) {
            v16bf kf0 = load_frag_b(Hb, kb + nn,      H, kc * 32, hi);
            v16bf kf1 = load_frag_b(Hb, kb + 16 + nn, H, kc * 32, hi);
            s0 = wmma_bf16(qf[kc], kf0, s0);
            s1 = wmma_bf16(qf[kc], kf1, s1);
        }
        // mask padded keys
        if (kb + nn >= N_NODES) {
#pragma unroll
            for (int j = 0; j < 8; ++j) s0[j] = -__builtin_inff();
        }
        if (kb + 16 + nn >= N_NODES) {
#pragma unroll
            for (int j = 0; j < 8; ++j) s1[j] = -__builtin_inff();
        }
        // row-wise max across the 16 lanes sharing a row group
        float tm[8];
#pragma unroll
        for (int j = 0; j < 8; ++j) tm[j] = fmaxf(s0[j], s1[j]);
#pragma unroll
        for (int off = 1; off < 16; off <<= 1) {
#pragma unroll
            for (int j = 0; j < 8; ++j)
                tm[j] = fmaxf(tm[j], __shfl_xor(tm[j], off, 32));
        }
        float alpha[8];
#pragma unroll
        for (int j = 0; j < 8; ++j) {
            float mn = fmaxf(m[j], tm[j]);
            alpha[j] = __expf(m[j] - mn);
            m[j] = mn;
        }
        float p0[8], p1[8], rs[8];
#pragma unroll
        for (int j = 0; j < 8; ++j) {
            p0[j] = __expf(s0[j] - m[j]);
            p1[j] = __expf(s1[j] - m[j]);
            rs[j] = p0[j] + p1[j];
        }
#pragma unroll
        for (int off = 1; off < 16; off <<= 1) {
#pragma unroll
            for (int j = 0; j < 8; ++j) rs[j] += __shfl_xor(rs[j], off, 32);
        }
#pragma unroll
        for (int j = 0; j < 8; ++j) l[j] = l[j] * alpha[j] + rs[j];
        // rescale O
#pragma unroll
        for (int ct = 0; ct < 8; ++ct)
#pragma unroll
            for (int j = 0; j < 8; ++j) O[ct][j] *= alpha[j];
        // transpose P (16x32) through wave-private LDS, reload as A-fragment
#pragma unroll
        for (int j = 0; j < 8; ++j) {
            int row = j + hi * 8;
            P[row * 32 + nn]      = (__bf16)p0[j];
            P[row * 32 + 16 + nn] = (__bf16)p1[j];
        }
        v16bf pf = load_frag_a(P, nn, 32, 0, hi);
        // O += P @ V  (V columns streamed from HbT, contiguous along keys)
#pragma unroll
        for (int ct = 0; ct < 8; ++ct) {
            v16bf vf = load_frag_b(HbT, half * 128 + ct * 16 + nn, NPAD, kb, hi);
            O[ct] = wmma_bf16(pf, vf, O[ct]);
        }
    }

    // epilogue: enc = O / l ; x = [h, h-enc, h*enc]
    float linv[8];
#pragma unroll
    for (int j = 0; j < 8; ++j) linv[j] = (l[j] > 0.0f) ? 1.0f / l[j] : 0.0f;
#pragma unroll
    for (int ct = 0; ct < 8; ++ct) {
        int col = half * 128 + ct * 16 + nn;
#pragma unroll
        for (int j = 0; j < 8; ++j) {
            int row = qrow + j + hi * 8;
            if (row < N_NODES) {
                float e = O[ct][j] * linv[j];
                float h = hidden[(size_t)row * H + col];
                __bf16* xp = Xb + (size_t)row * XDIM;
                xp[col]         = (__bf16)h;
                xp[H + col]     = (__bf16)(h - e);
                xp[2 * H + col] = (__bf16)(h * e);
            }
        }
    }
}

// ---------------------------------------------------------------------------
// K3: per-level A = [x_lvl | lh | rh]  (npad x 1280 bf16, zero padded)
// ---------------------------------------------------------------------------
__global__ __launch_bounds__(256) void build_A_kernel(
    const __bf16* __restrict__ Xb, const __bf16* __restrict__ hb_prev,
    __bf16* __restrict__ Ab, int offset, int nrows, int npad, int has_prev) {
    int idx = blockIdx.x * blockDim.x + threadIdx.x;
    if (idx >= npad * KCAT) return;
    int r = idx / KCAT, c = idx % KCAT;
    __bf16 v = (__bf16)0.0f;
    if (r < nrows) {
        if (c < XDIM) v = Xb[(size_t)(offset + r) * XDIM + c];
        else if (has_prev) {
            if (c < XDIM + MDIM) v = hb_prev[(size_t)(2 * r) * MDIM + (c - XDIM)];
            else                 v = hb_prev[(size_t)(2 * r + 1) * MDIM + (c - XDIM - MDIM)];
        }
    }
    Ab[idx] = v;
}

// ---------------------------------------------------------------------------
// K4: level GEMM  G = A(npad x 1280) @ Wcat(1280 x 1280), bf16 WMMA, f32 acc
// One wave = one 16x16 C tile. grid = (20, npad/16), 4 waves/block.
// ---------------------------------------------------------------------------
__global__ __launch_bounds__(128) void gemm_level_kernel(
    const __bf16* __restrict__ Ab, const __bf16* __restrict__ WcatT,
    float* __restrict__ G, int npad) {
    const int lane = threadIdx.x & 31;
    const int wave = threadIdx.x >> 5;
    const int ct = blockIdx.x * 4 + wave;   // 0..79
    const int rt = blockIdx.y;
    const int nn = lane & 15;
    const int hi = lane >> 4;
    v8f acc = vzero8();
#pragma unroll 4
    for (int kc = 0; kc < KCAT / 32; ++kc) {
        v16bf a = load_frag_a(Ab,    rt * 16 + nn, KCAT, kc * 32, hi);
        v16bf b = load_frag_b(WcatT, ct * 16 + nn, KCAT, kc * 32, hi);
        acc = wmma_bf16(a, b, acc);
    }
#pragma unroll
    for (int j = 0; j < 8; ++j)
        G[(size_t)(rt * 16 + j + hi * 8) * GDIM + ct * 16 + nn] = acc[j];
    (void)npad;
}

// ---------------------------------------------------------------------------
// K5: LSTM gates:  c = sig(i)tanh(u)+sig(fl)lc+sig(fr)rc ; h = sig(o)tanh(c)
// ---------------------------------------------------------------------------
__global__ __launch_bounds__(256) void gates_kernel(
    const float* __restrict__ G, const float* __restrict__ bias,
    const float* __restrict__ c_prev, float* __restrict__ c_out,
    float* __restrict__ h_out, __bf16* __restrict__ hb_out,
    int nrows, int has_prev) {
    int idx = blockIdx.x * blockDim.x + threadIdx.x;
    if (idx >= nrows * MDIM) return;
    int r = idx / MDIM, mc = idx % MDIM;
    const float* g = G + (size_t)r * GDIM;
    float gi = g[mc]            + bias[mc];
    float fl = g[MDIM + mc]     + bias[MDIM + mc];
    float fr = g[2 * MDIM + mc] + bias[2 * MDIM + mc];
    float go = g[3 * MDIM + mc] + bias[3 * MDIM + mc];
    float gu = g[4 * MDIM + mc] + bias[4 * MDIM + mc];
    float lc = 0.0f, rc = 0.0f;
    if (has_prev) {
        lc = c_prev[(size_t)(2 * r) * MDIM + mc];
        rc = c_prev[(size_t)(2 * r + 1) * MDIM + mc];
    }
    float si = 1.0f / (1.0f + __expf(-gi));
    float sl = 1.0f / (1.0f + __expf(-fl));
    float sr = 1.0f / (1.0f + __expf(-fr));
    float so = 1.0f / (1.0f + __expf(-go));
    float c  = si * tanhf(gu) + sl * lc + sr * rc;
    float h  = so * tanhf(c);
    c_out[idx]  = c;
    h_out[idx]  = h;
    hb_out[idx] = (__bf16)h;
}

// ---------------------------------------------------------------------------
extern "C" void kernel_launch(void* const* d_in, const int* in_sizes, int n_in,
                              void* d_out, int out_size, void* d_ws,
                              size_t ws_size, hipStream_t stream) {
    (void)in_sizes; (void)n_in; (void)out_size; (void)ws_size;
    const float* hidden = (const float*)d_in[0];
    const float* W      = (const float*)d_in[1];
    const float* Ul     = (const float*)d_in[2];
    const float* Ur     = (const float*)d_in[3];
    const float* bias   = (const float*)d_in[4];
    float* out = (float*)d_out;

    char* ws = (char*)d_ws;
    size_t off = 0;
    auto alloc = [&](size_t bytes) -> char* {
        char* p = ws + off;
        off += (bytes + 255) & ~(size_t)255;
        return p;
    };
    __bf16* Hb    = (__bf16*)alloc((size_t)NPAD * H * 2);
    __bf16* HbT   = (__bf16*)alloc((size_t)H * NPAD * 2);
    __bf16* Xb    = (__bf16*)alloc((size_t)NPAD * XDIM * 2);
    __bf16* WcatT = (__bf16*)alloc((size_t)GDIM * KCAT * 2);
    __bf16* Ab    = (__bf16*)alloc((size_t)4096 * KCAT * 2);
    float*  G     = (float*)alloc((size_t)4096 * GDIM * 4);
    float*  cA    = (float*)alloc((size_t)4096 * MDIM * 4);
    float*  cB    = (float*)alloc((size_t)4096 * MDIM * 4);
    __bf16* hbA   = (__bf16*)alloc((size_t)4096 * MDIM * 2);
    __bf16* hbB   = (__bf16*)alloc((size_t)4096 * MDIM * 2);

    cast_hidden_kernel<<<(NPAD * H + 255) / 256, 256, 0, stream>>>(hidden, Hb, HbT);
    build_wcatT_kernel<<<(GDIM * KCAT + 255) / 256, 256, 0, stream>>>(W, Ul, Ur, WcatT);
    attn_kernel<<<256, 128, 0, stream>>>(hidden, Hb, HbT, Xb);

    int offset = 0, n = 4096, level = 0;
    float*  cprev = cB;  float*  ccur = cA;
    __bf16* hprev = hbB; __bf16* hcur = hbA;
    while (n >= 1) {
        int npad = (n + 15) & ~15;
        int totA = npad * KCAT;
        build_A_kernel<<<(totA + 255) / 256, 256, 0, stream>>>(
            Xb, hprev, Ab, offset, n, npad, level > 0 ? 1 : 0);
        dim3 gg(GDIM / 16 / 4, npad / 16);
        gemm_level_kernel<<<gg, 128, 0, stream>>>(Ab, WcatT, G, npad);
        int totg = n * MDIM;
        gates_kernel<<<(totg + 255) / 256, 256, 0, stream>>>(
            G, bias, cprev, ccur, out + (size_t)offset * MDIM, hcur, n,
            level > 0 ? 1 : 0);
        float*  tc = cprev; cprev = ccur; ccur = tc;
        __bf16* th = hprev; hprev = hcur; hcur = th;
        offset += n;
        n >>= 1;
        ++level;
    }
}
